// Attn_45947560132882
// MI455X (gfx1250) — matrix operand order repaired and checked by reading, not compiled, not run on hardware
//
#include <hip/hip_runtime.h>

#define Bb 32
#define Ss 4096
#define Hh 512

typedef __attribute__((ext_vector_type(16))) __bf16 v16bf;
typedef __attribute__((ext_vector_type(8)))  float  v8f;
typedef unsigned short u16;
typedef unsigned int   u32;

#if __has_builtin(__builtin_amdgcn_tanhf)
#define FAST_TANH(x) __builtin_amdgcn_tanhf(x)
#else
#define FAST_TANH(x) tanhf(x)
#endif

__device__ __forceinline__ u32 pk_bf16(float a, float b) {
  u32 ua = __float_as_uint(a), ub = __float_as_uint(b);
  ua = (ua + 0x7FFFu + ((ua >> 16) & 1u)) >> 16;
  ub = (ub + 0x7FFFu + ((ub >> 16) & 1u)) >> 16;
  return ua | (ub << 16);
}
__device__ __forceinline__ u16 f2bf(float a) {
  u32 x = __float_as_uint(a);
  return (u16)((x + 0x7FFFu + ((x >> 16) & 1u)) >> 16);
}

// Async global->LDS copy of 16 bytes per lane (CDNA5, ASYNCcnt-tracked).
__device__ __forceinline__ void async_copy16(u32 lds_off, const void* gptr) {
  asm volatile("global_load_async_to_lds_b128 %0, %1, off"
               :: "v"(lds_off), "v"((unsigned long long)(uintptr_t)gptr)
               : "memory");
}
__device__ __forceinline__ void wait_async0() {
#if __has_builtin(__builtin_amdgcn_s_wait_asynccnt)
  __builtin_amdgcn_s_wait_asynccnt(0);
#else
  asm volatile("s_wait_asynccnt 0x0" ::: "memory");
#endif
}

// ---------------------------------------------------------------------------
// Pack W1 = W_w[:, 0:H] into bf16 WMMA-B tiles, grouped so each n-group
// (4 consecutive n-tiles = 64 cols) is a contiguous 64 KB block:
//   tile_linear = (nt>>2)*64 + kt*4 + (nt&3)
// Within a tile (K=32 x N=16): lane = n + 16*((k&8)!=0), slot = ((k&16)?8:0)+(k&7).
// ---------------------------------------------------------------------------
__global__ __launch_bounds__(512) void pack_w1_kernel(const float* __restrict__ Ww,
                                                      u16* __restrict__ w1p) {
  int flat = blockIdx.x * 512 + threadIdx.x;   // 0 .. 512*512-1
  int ng = flat >> 9;        // output feature n (row of W_w)
  int kg = flat & 511;       // k within first H columns
  float v = Ww[(size_t)ng * (2 * Hh) + kg];
  int nt = ng >> 4, n = ng & 15;
  int kt = kg >> 5, kk = kg & 31;
  int lane = n + ((kk & 8) ? 16 : 0);
  int slot = ((kk & 16) ? 8 : 0) + (kk & 7);
  size_t tile = (size_t)(nt >> 2) * 64 + kt * 4 + (nt & 3);
  w1p[((tile * 32 + lane) << 4) + slot] = f2bf(v);
}

// hb[b,h] = W_b[h] + sum_k hidden[b,k] * W_w[h, H + k]
__global__ __launch_bounds__(256) void hb_kernel(const float* __restrict__ hid,
                                                 const float* __restrict__ Ww,
                                                 const float* __restrict__ Wb,
                                                 float* __restrict__ hb) {
  int flat = blockIdx.x * 256 + threadIdx.x;   // 0 .. B*H-1
  int b = flat >> 9, h = flat & 511;
  const float* hr = hid + (size_t)b * Hh;
  const float* wr = Ww + (size_t)h * (2 * Hh) + Hh;
  float acc = Wb[h];
  for (int k = 0; k < Hh; ++k) acc += hr[k] * wr[k];
  hb[flat] = acc;
}

// u[h] = sum_g V_w[g,h];  c = sum(V_b)
__global__ __launch_bounds__(512) void uc_kernel(const float* __restrict__ Vw,
                                                 const float* __restrict__ Vb,
                                                 float* __restrict__ u,
                                                 float* __restrict__ cout) {
  __shared__ float red[512];
  int h = threadIdx.x;
  float acc = 0.f;
  for (int g = 0; g < Hh; ++g) acc += Vw[(size_t)g * Hh + h];
  u[h] = acc;
  red[h] = Vb[h];
  __syncthreads();
  for (int s2 = 256; s2 > 0; s2 >>= 1) {
    if (h < s2) red[h] += red[h + s2];
    __syncthreads();
  }
  if (h == 0) *cout = red[0];
}

// ---------------------------------------------------------------------------
// Main kernel: 128 rows/block, 8 waves x 16-row strips.
//  - A staged once in LDS (bf16 WMMA-A layout), then hoisted to 16 register
//    fragments per wave.
//  - B streamed n-group by n-group into a double-buffered 64 KB LDS slab via
//    global_load_async_to_lds_b128, shared by all 8 waves (8x less L2 traffic).
//  - Accumulator tiles consumed in registers: sc += tanh(pre+hb)*u, then a
//    shfl_xor butterfly produces one score per row.
// ---------------------------------------------------------------------------
__global__ __launch_bounds__(256) void scores_kernel(
    const float* __restrict__ inputs, const u16* __restrict__ w1p,
    const float* __restrict__ hb, const float* __restrict__ u,
    const float* __restrict__ cptr, float* __restrict__ scores) {
  __shared__ u16 lds_a[128 * 512];      // 128 KB : A, WMMA-A packed bf16
  __shared__ u16 lds_b[2][4096 * 8];    // 2 x 64 KB : B double buffer

  const int tid = threadIdx.x;
  const int m0  = blockIdx.x * 128;     // all 128 rows share batch index b
  const int b   = m0 >> 12;

  const char* w1bytes = (const char*)w1p;

  // ---- kick off async B loads for n-group 0 (overlaps the A staging below)
  {
    u32 dst = (u32)(uintptr_t)(&lds_b[0][0]);
#pragma unroll
    for (int i = 0; i < 16; ++i) {
      u32 idx = (u32)(i * 256 + tid);   // 16B chunk index, 4096 chunks = 64 KB
      async_copy16(dst + idx * 16, w1bytes + (size_t)idx * 16);
    }
  }

  // ---- stage + convert A: 8 consecutive k -> 16B contiguous LDS store.
  // A-layout (16x32 bf16): lane = m + 16*((k&8)!=0); slot = ((k&16)?8:0)+(k&7)
  for (int g = tid; g < 128 * 64; g += 256) {
    int mm = g >> 6;                    // row in block 0..127
    int k  = (g & 63) << 3;             // k group base 0..504
    const float* src = inputs + (size_t)(m0 + mm) * Hh + k;
    float4 f0 = *(const float4*)(src);
    float4 f1 = *(const float4*)(src + 4);
    uint4 d;
    d.x = pk_bf16(f0.x, f0.y); d.y = pk_bf16(f0.z, f0.w);
    d.z = pk_bf16(f1.x, f1.y); d.w = pk_bf16(f1.z, f1.w);
    int m_tile = mm >> 4, m = mm & 15;
    int kt = k >> 5, kk = k & 31;
    int lane = m + ((kk & 8) ? 16 : 0);
    int base = (kk & 16) ? 8 : 0;
    *(uint4*)&lds_a[(((m_tile * 16 + kt) * 32 + lane) << 4) + base] = d;
  }
  wait_async0();
  __syncthreads();

  const int wave = tid >> 5;            // 16-row strip
  const int lane = tid & 31;
  const int nlo  = lane & 15;
  const float* hbr = hb + (size_t)b * Hh;

  // Hoist this wave's 16 A fragments into registers (128 VGPRs).
  const v16bf* Ap = (const v16bf*)lds_a;
  v16bf afrag[16];
#pragma unroll
  for (int kt = 0; kt < 16; ++kt)
    afrag[kt] = Ap[(wave * 16 + kt) * 32 + lane];

  float sc[8] = {0.f, 0.f, 0.f, 0.f, 0.f, 0.f, 0.f, 0.f};

  for (int ntg = 0; ntg < 8; ++ntg) {
    const int cur = ntg & 1;

    // Prefetch next n-group's B slab into the other buffer (async).
    if (ntg < 7) {
      u32 dst = (u32)(uintptr_t)(&lds_b[cur ^ 1][0]);
      const char* src = w1bytes + (size_t)(ntg + 1) * 65536;
#pragma unroll
      for (int i = 0; i < 16; ++i) {
        u32 idx = (u32)(i * 256 + tid);
        async_copy16(dst + idx * 16, src + (size_t)idx * 16);
      }
    }

    const v16bf* Bp = (const v16bf*)&lds_b[cur][0];
    v8f acc[4] = {};
#pragma unroll
    for (int kt = 0; kt < 16; ++kt) {
#pragma unroll
      for (int j = 0; j < 4; ++j) {
        v16bf bm = Bp[(kt * 4 + j) * 32 + lane];
        acc[j] = __builtin_amdgcn_wmma_f32_16x16x32_bf16(
            false, afrag[kt], false, bm, (short)0, acc[j], false, false);
      }
    }

    // Consume: C layout -> lane covers col n = nlo, rows r + 8*(lane>=16)
#pragma unroll
    for (int j = 0; j < 4; ++j) {
      int ng = (ntg * 4 + j) * 16 + nlo;
      float hbv = hbr[ng];
      float uv  = u[ng];
#pragma unroll
      for (int r = 0; r < 8; ++r)
        sc[r] += FAST_TANH(acc[j][r] + hbv) * uv;
    }

    wait_async0();
    __syncthreads();
  }

  float c = *cptr;
  int mrow = m0 + wave * 16;
#pragma unroll
  for (int r = 0; r < 8; ++r) {
    float v = sc[r];
    v += __shfl_xor(v, 1, 32);
    v += __shfl_xor(v, 2, 32);
    v += __shfl_xor(v, 4, 32);
    v += __shfl_xor(v, 8, 32);   // reduce within each 16-lane half
    if (lane == 0)  scores[mrow + r]     = v + c;   // rows 0..7
    if (lane == 16) scores[mrow + 8 + r] = v + c;   // rows 8..15
  }
}

// In-place softmax over S per batch.
__global__ __launch_bounds__(256) void softmax_kernel(float* __restrict__ attn) {
  __shared__ float red[256];
  const int b = blockIdx.x, tid = threadIdx.x;
  float* p = attn + (size_t)b * Ss;
  float loc[16];
  float mx = -3.402823466e38f;
#pragma unroll
  for (int i = 0; i < 16; ++i) { loc[i] = p[tid + i * 256]; mx = fmaxf(mx, loc[i]); }
  red[tid] = mx; __syncthreads();
  for (int s2 = 128; s2 > 0; s2 >>= 1) {
    if (tid < s2) red[tid] = fmaxf(red[tid], red[tid + s2]);
    __syncthreads();
  }
  mx = red[0]; __syncthreads();
  float sum = 0.f;
#pragma unroll
  for (int i = 0; i < 16; ++i) { loc[i] = __expf(loc[i] - mx); sum += loc[i]; }
  red[tid] = sum; __syncthreads();
  for (int s2 = 128; s2 > 0; s2 >>= 1) {
    if (tid < s2) red[tid] += red[tid + s2];
    __syncthreads();
  }
  float inv = 1.f / red[0];
#pragma unroll
  for (int i = 0; i < 16; ++i) p[tid + i * 256] = loc[i] * inv;
}

// res = attn * inputs, float4 streams.
__global__ __launch_bounds__(256) void res_kernel(const float* __restrict__ inputs,
                                                  const float* __restrict__ attn,
                                                  float* __restrict__ out) {
  size_t i = (size_t)blockIdx.x * 256 + threadIdx.x;   // float4 index
  size_t row = i >> 7;                                 // H/4 = 128 vec4 per row
  float a = attn[row];
  float4 v = ((const float4*)inputs)[i];
  v.x *= a; v.y *= a; v.z *= a; v.w *= a;
  ((float4*)out)[i] = v;
}

extern "C" void kernel_launch(void* const* d_in, const int* in_sizes, int n_in,
                              void* d_out, int out_size, void* d_ws, size_t ws_size,
                              hipStream_t stream) {
  (void)in_sizes; (void)n_in; (void)out_size; (void)ws_size;
  const float* inputs = (const float*)d_in[0];
  const float* hidden = (const float*)d_in[1];   // (1,B,H)
  const float* Ww     = (const float*)d_in[2];   // (H, 2H)
  const float* Wb     = (const float*)d_in[3];   // (H,)
  const float* Vw     = (const float*)d_in[4];   // (H, H)
  const float* Vb     = (const float*)d_in[5];   // (H,)

  char* ws = (char*)d_ws;
  u16*   w1p = (u16*)ws;                                    // 512 KB packed bf16 W1
  float* hb  = (float*)(ws + (512u << 10));                 // 64 KB
  float* u   = (float*)(ws + (512u << 10) + (64u << 10));   // 2 KB
  float* c   = (float*)(ws + (512u << 10) + (64u << 10) + 2048);

  float* out  = (float*)d_out;                       // res: B*S*H
  float* attn = out + (size_t)Bb * Ss * Hh;          // attn: B*S (scores scratch)

  pack_w1_kernel<<<512, 512, 0, stream>>>(Ww, w1p);
  hb_kernel<<<(Bb * Hh) / 256, 256, 0, stream>>>(hidden, Ww, Wb, hb);
  uc_kernel<<<1, 512, 0, stream>>>(Vw, Vb, u, c);
  scores_kernel<<<(Bb * Ss) / 128, 256, 0, stream>>>(inputs, w1p, hb, u, c, attn);
  softmax_kernel<<<Bb, 256, 0, stream>>>(attn);
  res_kernel<<<((size_t)Bb * Ss * Hh / 4) / 256, 256, 0, stream>>>(inputs, attn, out);
}
